// ResTAttention_46926812676790
// MI455X (gfx1250) — matrix operand order repaired and checked by reading, not compiled, not run on hardware
//
#include <hip/hip_runtime.h>
#include <hip/hip_bf16.h>
#include <stdint.h>

// ---------------------------------------------------------------------------
// ResTAttention fused pipeline for MI455X (gfx1250, wave32, WMMA f16->f32)
//   B=8 C=256 H=W=48 (N=2304)  NH=8 HD=32  SR=2 -> M=576
// Key fusions:
//   * InstanceNorm mean == 1/M exactly (softmax rows sum to 1); var from
//     online sum(p^2)/l^2 during flash softmax -> norm folds into a
//     per-column affine applied inside the projection GEMM A-load.
//   * head-mix (w_tc) folds into the QK^T contraction as per-K-chunk scalar
//     scaling of q (logits_g = sum_c (w_tc[g,c>>5]*q[n,c]) * k[m,c]).
//   * attn tensor (340MB) never materialized: flash per (b,g,n-tile) wave.
// ---------------------------------------------------------------------------

typedef _Float16 half_t;
typedef __attribute__((ext_vector_type(16))) _Float16 v16h;
typedef __attribute__((ext_vector_type(8)))  _Float16 v8h;
typedef __attribute__((ext_vector_type(8)))  float    v8f;
typedef __attribute__((ext_vector_type(4)))  float    v4f;

#define B_    8
#define C_    256
#define H_    48
#define W_    48
#define N_    2304
#define NH_   8
#define HD_   32
#define M_    576
#define MH_   24
#define EPS_  1e-5f
#define SCALE_ 0.17677669529663687f   /* 32^-0.5 */

// ---------------- WMMA helpers (A/B operands use contiguous K = 16h + s) -------------------------

__device__ __forceinline__ v8f wmma16(v16h a, v16h b, v8f c) {
  return __builtin_amdgcn_wmma_f32_16x16x32_f16(false, a, false, b, (short)0, c,
                                                false, false);
}

// A-operand 16x32 f16 from row-major [row][K] (ld in halves).
// lane<16: row=lane,   K = 0..15
// lane>=16: row=lane-16, K = 16..31
__device__ __forceinline__ v16h load_a32(const half_t* __restrict__ base,
                                         int ld, int lane) {
  const half_t* p = base + (size_t)(lane & 15) * ld + ((lane & 16) ? 16 : 0);
  union { v16h v; v8h h[2]; } u;
  u.h[0] = *(const v8h*)p;
  u.h[1] = *(const v8h*)(p + 8);
  return u.v;
}

// B-operand 32x16 f16 from "Bt" layout [col][K] (K contiguous, ld in halves).
// lane<16: col=lane, K=0..15 ; lane>=16: col=lane-16, K=16..31
__device__ __forceinline__ v16h load_b32(const half_t* __restrict__ base,
                                         int ld, int lane) {
  const half_t* p = base + (size_t)(lane & 15) * ld + ((lane & 16) ? 16 : 0);
  union { v16h v; v8h h[2]; } u;
  u.h[0] = *(const v8h*)p;
  u.h[1] = *(const v8h*)(p + 8);
  return u.v;
}

// C 16x16 f32 -> f16 store to row-major [row][col]; out pre-offset to tile origin.
__device__ __forceinline__ void store_c_f16(half_t* __restrict__ out, int ld,
                                            int lane, v8f acc) {
  half_t* p = out + (size_t)((lane & 16) ? 8 : 0) * ld + (lane & 15);
#pragma unroll
  for (int r = 0; r < 8; ++r) p[(size_t)r * ld] = (half_t)acc[r];
}

// ---------------- Kernels ---------------------------------------------------

// Weights: transpose to [out][in] f16; fold SCALE into w_q.
__global__ void k_prep_weights(const float* __restrict__ w_q,
                               const float* __restrict__ w_kv,
                               const float* __restrict__ w_proj,
                               half_t* __restrict__ wqT,
                               half_t* __restrict__ wkvT,
                               half_t* __restrict__ wpT) {
  int idx = blockIdx.x * 256 + threadIdx.x;        // 262144 total
  if (idx < 65536) {
    int oc = idx >> 8, ic = idx & 255;
    wqT[idx] = (half_t)(w_q[ic * 256 + oc] * SCALE_);
  } else if (idx < 65536 + 131072) {
    int j = idx - 65536;
    int oc = j >> 8, ic = j & 255;
    wkvT[j] = (half_t)w_kv[ic * 512 + oc];
  } else {
    int j = idx - 196608;
    int oc = j >> 8, ic = j & 255;
    wpT[j] = (half_t)w_proj[ic * 256 + oc];
  }
}

// x [B,C,N] f32 -> xt16 [B,N,C] f16 (LDS tiled transpose)
__global__ void k_transpose_x(const float* __restrict__ x,
                              half_t* __restrict__ xt) {
  __shared__ float tile[32][33];
  int b = blockIdx.z, c0 = blockIdx.y * 32, n0 = blockIdx.x * 32;
  int tx = threadIdx.x, ty = threadIdx.y;          // block (32,8)
  const float* src = x + ((size_t)(b * C_ + c0)) * N_ + n0;
  for (int i = ty; i < 32; i += 8) tile[i][tx] = src[(size_t)i * N_ + tx];
  __syncthreads();
  half_t* dst = xt + ((size_t)(b * N_ + n0)) * C_ + c0;
  for (int i = ty; i < 32; i += 8)
    dst[(size_t)i * C_ + tx] = (half_t)tile[tx][i];
}

// q16[b*n, c] = xt16 @ (w_q*SCALE)   rows 18432, cols 256, K 256
__global__ void __launch_bounds__(128) k_qproj(const half_t* __restrict__ xt,
                                               const half_t* __restrict__ wqT,
                                               half_t* __restrict__ q16) {
  int lane = threadIdx.x & 31, wave = threadIdx.x >> 5;
  int row0 = blockIdx.x * 16;
  int col0 = (blockIdx.y * 4 + wave) * 16;
  const half_t* A  = xt  + (size_t)row0 * C_;
  const half_t* Bt = wqT + (size_t)col0 * C_;
  v8f acc;
#pragma unroll
  for (int i = 0; i < 8; ++i) acc[i] = 0.f;
#pragma unroll
  for (int kc = 0; kc < 8; ++kc)
    acc = wmma16(load_a32(A + kc * 32, C_, lane),
                 load_b32(Bt + kc * 32, C_, lane), acc);
  store_c_f16(q16 + (size_t)row0 * C_ + col0, C_, lane, acc);
}

// depthwise 3x3 s2 conv + b_sr + LayerNorm(C) -> xk16 [b*m, c] f16
__global__ void k_sr_ln(const float* __restrict__ x,
                        const float* __restrict__ w_sr,
                        const float* __restrict__ b_sr,
                        const float* __restrict__ g_ln,
                        const float* __restrict__ b_ln,
                        half_t* __restrict__ xk) {
  __shared__ float red[2][8];
  int bm = blockIdx.x;
  int b = bm / M_, m = bm % M_;
  int c = threadIdx.x;
  int oh = m / MH_, ow = m % MH_;
  const float* xp = x + ((size_t)(b * C_ + c)) * N_;
  const float* wp = w_sr + c * 9;
  float y = 0.f;
#pragma unroll
  for (int kh = 0; kh < 3; ++kh) {
    int ih = oh * 2 - 1 + kh;
    if (ih < 0 || ih >= H_) continue;
#pragma unroll
    for (int kw = 0; kw < 3; ++kw) {
      int iw = ow * 2 - 1 + kw;
      if (iw < 0 || iw >= W_) continue;
      y += wp[kh * 3 + kw] * xp[ih * W_ + iw];
    }
  }
  y += b_sr[c];
  float v1 = y, v2 = y * y;
#pragma unroll
  for (int off = 16; off; off >>= 1) {
    v1 += __shfl_xor(v1, off, 32);
    v2 += __shfl_xor(v2, off, 32);
  }
  int wv = c >> 5, ln = c & 31;
  if (ln == 0) { red[0][wv] = v1; red[1][wv] = v2; }
  __syncthreads();
  if (wv == 0) {
    float a = (ln < 8) ? red[0][ln] : 0.f;
    float q = (ln < 8) ? red[1][ln] : 0.f;
#pragma unroll
    for (int off = 4; off; off >>= 1) {
      a += __shfl_xor(a, off, 32);
      q += __shfl_xor(q, off, 32);
    }
    if (ln == 0) { red[0][0] = a; red[1][0] = q; }
  }
  __syncthreads();
  float mu  = red[0][0] * (1.f / C_);
  float var = red[1][0] * (1.f / C_) - mu * mu;
  float rsv = rsqrtf(var + EPS_);
  xk[(size_t)bm * C_ + c] = (half_t)((y - mu) * rsv * g_ln[c] + b_ln[c]);
}

// kv = xk16 @ w_kv : cols 0..255 -> k16[b*m,c] ; cols 256..511 -> vT16[b,h,d,m]
__global__ void __launch_bounds__(128) k_kvproj(const half_t* __restrict__ xk,
                                                const half_t* __restrict__ wkvT,
                                                half_t* __restrict__ k16,
                                                half_t* __restrict__ vT) {
  int lane = threadIdx.x & 31, wave = threadIdx.x >> 5;
  int row0 = blockIdx.x * 16;                        // 288 row tiles
  int col0 = (blockIdx.y * 4 + wave) * 16;           // 32 col tiles
  const half_t* A  = xk   + (size_t)row0 * C_;
  const half_t* Bt = wkvT + (size_t)col0 * C_;
  v8f acc;
#pragma unroll
  for (int i = 0; i < 8; ++i) acc[i] = 0.f;
#pragma unroll
  for (int kc = 0; kc < 8; ++kc)
    acc = wmma16(load_a32(A + kc * 32, C_, lane),
                 load_b32(Bt + kc * 32, C_, lane), acc);
  if (col0 < C_) {
    store_c_f16(k16 + (size_t)row0 * C_ + col0, C_, lane, acc);
  } else {
    int cp = col0 - C_;
    int h = cp >> 5;                                  // head
    int d = (cp & 31) + (lane & 15);                  // fixed per lane
    int b = blockIdx.x / 36;
    int m0 = (blockIdx.x % 36) * 16 + ((lane & 16) ? 8 : 0);
    v8h pk;
#pragma unroll
    for (int r = 0; r < 8; ++r) pk[r] = (half_t)acc[r];
    *(v8h*)(vT + ((size_t)(b * NH_ + h) * HD_ + d) * M_ + m0) = pk;
  }
}

// column sums of v per (b,h,d)  (for InstanceNorm mean folding)
__global__ void k_vsum(const half_t* __restrict__ vT,
                       float* __restrict__ colsum) {
  int b = blockIdx.x, t = threadIdx.x;
  int g = t >> 5, d = t & 31;
  const half_t* p = vT + ((size_t)(b * NH_ + g) * HD_ + d) * M_;
  float s = 0.f;
  for (int m = 0; m < M_; ++m) s += (float)p[m];
  colsum[(b * NH_ + g) * HD_ + d] = s;
}

// Flash attention per (b, g, n-tile-16): logits^T WMMA (K=256 w/ folded head
// mix), online softmax + sum(p^2), PV as O^T = v^T @ attn^T WMMA.
__global__ void __launch_bounds__(128, 1)
k_attn(const half_t* __restrict__ q16, const half_t* __restrict__ k16,
       const half_t* __restrict__ vT, const float* __restrict__ w_tc,
       const float* __restrict__ b_tc, half_t* __restrict__ outU,
       float* __restrict__ s2part) {
  int lane = threadIdx.x & 31, wave = threadIdx.x >> 5;
  int bg = blockIdx.x;                 // 64
  int b = bg >> 3, g = bg & 7;
  int nt = blockIdx.y * 4 + wave;      // 144 n-tiles
  int n0 = nt * 16;
  float btc = b_tc[g];

  // q~^T B-operand chunks: q16 row scaled by w_tc[g, h=kc]
  v16h qb[8];
  const half_t* qrow = q16 + ((size_t)(b * N_) + n0 + (lane & 15)) * C_;
#pragma unroll
  for (int kc = 0; kc < 8; ++kc) {
    const half_t* p = qrow + kc * 32 + ((lane & 16) ? 16 : 0);
    union { v16h v; v8h h[2]; } u;
    u.h[0] = *(const v8h*)p;
    u.h[1] = *(const v8h*)(p + 8);
    _Float16 s = (_Float16)w_tc[g * NH_ + kc];
#pragma unroll
    for (int i = 0; i < 16; ++i) u.v[i] *= s;
    qb[kc] = u.v;
  }

  float mrow = -1e30f, lrow = 0.f, s2row = 0.f;
  v8f o1, o2;                           // O^T tiles: d0..15 x n, d16..31 x n
#pragma unroll
  for (int i = 0; i < 8; ++i) { o1[i] = 0.f; o2[i] = 0.f; }

  const half_t* vbase0 = vT + ((size_t)(b * NH_ + g) * HD_) * M_;

#pragma unroll 1
  for (int mc = 0; mc < 18; ++mc) {     // 32 m per chunk
    int m0 = mc * 32;
    v8f acc1, acc2;                     // logits^T C tiles (m x n)
#pragma unroll
    for (int i = 0; i < 8; ++i) { acc1[i] = btc; acc2[i] = btc; }
    const half_t* kb1 = k16 + ((size_t)(b * M_) + m0) * C_;
    const half_t* kb2 = kb1 + 16 * C_;
#pragma unroll
    for (int kc = 0; kc < 8; ++kc)
      acc1 = wmma16(load_a32(kb1 + kc * 32, C_, lane), qb[kc], acc1);
#pragma unroll
    for (int kc = 0; kc < 8; ++kc)
      acc2 = wmma16(load_a32(kb2 + kc * 32, C_, lane), qb[kc], acc2);

    // ---- online softmax over this 32-m chunk (row n == lane&15) ----
    float cmax = acc1[0];
#pragma unroll
    for (int i = 1; i < 8; ++i) cmax = fmaxf(cmax, acc1[i]);
#pragma unroll
    for (int i = 0; i < 8; ++i) cmax = fmaxf(cmax, acc2[i]);
    cmax = fmaxf(cmax, __shfl_xor(cmax, 16, 32));
    float mnew  = fmaxf(mrow, cmax);
    float alpha = __expf(mrow - mnew);
    lrow  *= alpha;
    s2row *= alpha * alpha;
#pragma unroll
    for (int i = 0; i < 8; ++i) { o1[i] *= alpha; o2[i] *= alpha; }

    union { v8h h; int w[4]; } t1, t2;
    float ps = 0.f, p2 = 0.f;
#pragma unroll
    for (int i = 0; i < 8; ++i) {
      float p = __expf(acc1[i] - mnew);
      ps += p; p2 += p * p; t1.h[i] = (half_t)p;
    }
#pragma unroll
    for (int i = 0; i < 8; ++i) {
      float p = __expf(acc2[i] - mnew);
      ps += p; p2 += p * p; t2.h[i] = (half_t)p;
    }
    lrow += ps; s2row += p2;

    // assemble attn^T B-operand (K = 32 m, N = 16 n) via lane-pair shuffles
    int o1w[4], o2w[4];
#pragma unroll
    for (int j = 0; j < 4; ++j) {
      o1w[j] = __shfl_xor(t1.w[j], 16, 32);
      o2w[j] = __shfl_xor(t2.w[j], 16, 32);
    }
    union { v16h v; int w[8]; } pb;
    bool lo = (lane & 16) == 0;
#pragma unroll
    for (int j = 0; j < 4; ++j) {
      pb.w[j]     = lo ? t1.w[j] : o2w[j];
      pb.w[4 + j] = lo ? o1w[j]  : t2.w[j];
    }

    // O^T += v^T(d x m) * attn^T(m x n)
    {
      const half_t* p = vbase0 + (size_t)(lane & 15) * M_ + m0 +
                        ((lane & 16) ? 16 : 0);
      union { v16h v; v8h h[2]; } ua;
      ua.h[0] = *(const v8h*)p; ua.h[1] = *(const v8h*)(p + 8);
      o1 = wmma16(ua.v, pb.v, o1);
    }
    {
      const half_t* p = vbase0 + (size_t)(16 + (lane & 15)) * M_ + m0 +
                        ((lane & 16) ? 16 : 0);
      union { v16h v; v8h h[2]; } ua;
      ua.h[0] = *(const v8h*)p; ua.h[1] = *(const v8h*)(p + 8);
      o2 = wmma16(ua.v, pb.v, o2);
    }
    mrow = mnew;
  }

  float ltot  = lrow  + __shfl_xor(lrow, 16, 32);
  float s2tot = s2row + __shfl_xor(s2row, 16, 32);
  float inv = 1.f / ltot;

  // store unnormalized O (== softmax @ v); O^T C-layout lands as A-layout rows
  half_t* op = outU + ((size_t)(b * N_) + n0 + (lane & 15)) * C_ + g * HD_ +
               ((lane & 16) ? 8 : 0);
  v8h s1, s2h;
#pragma unroll
  for (int i = 0; i < 8; ++i) {
    s1[i]  = (half_t)(o1[i] * inv);
    s2h[i] = (half_t)(o2[i] * inv);
  }
  *(v8h*)op        = s1;
  *(v8h*)(op + 16) = s2h;

  // deterministic partial sum of attn^2 per (b,g): one slot per n-tile
  float contrib = (lane & 16) ? 0.f : s2tot * inv * inv;
#pragma unroll
  for (int off = 16; off; off >>= 1) contrib += __shfl_xor(contrib, off, 32);
  if (lane == 0) s2part[(size_t)bg * 144 + nt] = contrib;
}

// rs[b,g] = rsqrt(var+eps); biascol[b,c] = -rs * (1/M) * colsum_v
__global__ void k_norm(const float* __restrict__ s2part,
                       const float* __restrict__ colsum,
                       float* __restrict__ rs, float* __restrict__ biascol) {
  __shared__ float rr[8];
  int b = blockIdx.x, t = threadIdx.x;
  int g = t >> 5, d = t & 31;
  if (t < 8) {
    float s = 0.f;
    for (int i = 0; i < 144; ++i) s += s2part[(size_t)(b * NH_ + t) * 144 + i];
    float meanSq = s / ((float)N_ * (float)M_);
    float mu = 1.f / (float)M_;
    rr[t] = rsqrtf(meanSq - mu * mu + EPS_);
  }
  __syncthreads();
  float r = rr[g];
  if ((t & 31) == 0) rs[b * NH_ + g] = r;
  biascol[b * C_ + t] = -r * (1.f / (float)M_) * colsum[(b * NH_ + g) * HD_ + d];
}

// out = (rs*(outU - mu*colsum)) @ w_proj + b_proj, stored transposed [B,C,N] f32
__global__ void __launch_bounds__(128) k_proj(const half_t* __restrict__ outU,
                                              const half_t* __restrict__ wpT,
                                              const float* __restrict__ rs,
                                              const float* __restrict__ biascol,
                                              const float* __restrict__ b_proj,
                                              float* __restrict__ out) {
  int lane = threadIdx.x & 31, wave = threadIdx.x >> 5;
  int row0 = blockIdx.x * 16;                        // 1152 tiles, no b straddle
  int col0 = (blockIdx.y * 4 + wave) * 16;
  int b = row0 / N_;
  v8f acc;
  float bp = b_proj[col0 + (lane & 15)];
#pragma unroll
  for (int i = 0; i < 8; ++i) acc[i] = bp;
  const half_t* Bt = wpT + (size_t)col0 * C_;
#pragma unroll
  for (int kc = 0; kc < 8; ++kc) {
    float rsc = rs[b * NH_ + kc];                    // head == kc for this chunk
    const half_t* p = outU + ((size_t)row0 + (lane & 15)) * C_ + kc * 32 +
                      ((lane & 16) ? 16 : 0);
    const float* bb = biascol + b * C_ + kc * 32 + ((lane & 16) ? 16 : 0);
    v8h h0 = *(const v8h*)p, h1 = *(const v8h*)(p + 8);
    union { v16h v; v8h h[2]; } a;
#pragma unroll
    for (int i = 0; i < 8; ++i) {
      a.h[0][i] = (half_t)(rsc * (float)h0[i] + bb[i]);
      a.h[1][i] = (half_t)(rsc * (float)h1[i] + bb[8 + i]);
    }
    acc = wmma16(a.v, load_b32(Bt + kc * 32, C_, lane), acc);
  }
  // C: lane=cout col, VGPR=row n -> contiguous n store to [B,C,N]
  int cout = col0 + (lane & 15);
  int nrel = (row0 % N_) + ((lane & 16) ? 8 : 0);
  float* op = out + ((size_t)(b * C_) + cout) * N_ + nrel;
  union { v8f v; v4f q[2]; } s; s.v = acc;
  *(v4f*)op       = s.q[0];
  *(v4f*)(op + 4) = s.q[1];
}

// ---------------------------------------------------------------------------

extern "C" void kernel_launch(void* const* d_in, const int* in_sizes, int n_in,
                              void* d_out, int out_size, void* d_ws,
                              size_t ws_size, hipStream_t stream) {
  const float* x      = (const float*)d_in[0];
  const float* w_q    = (const float*)d_in[1];
  const float* w_kv   = (const float*)d_in[2];
  const float* w_sr   = (const float*)d_in[3];
  const float* b_sr   = (const float*)d_in[4];
  const float* g_ln   = (const float*)d_in[5];
  const float* b_ln   = (const float*)d_in[6];
  const float* w_tc   = (const float*)d_in[7];
  const float* b_tc   = (const float*)d_in[8];
  const float* w_proj = (const float*)d_in[9];
  const float* b_proj = (const float*)d_in[10];
  float* out = (float*)d_out;

  uint8_t* w = (uint8_t*)d_ws;
  auto take = [&](size_t bytes) {
    uint8_t* p = w;
    w += (bytes + 255) & ~(size_t)255;
    return p;
  };
  half_t* wqT     = (half_t*)take((size_t)C_ * C_ * 2);
  half_t* wkvT    = (half_t*)take((size_t)C_ * 2 * C_ * 2);
  half_t* wpT     = (half_t*)take((size_t)C_ * C_ * 2);
  half_t* xt16    = (half_t*)take((size_t)B_ * N_ * C_ * 2);
  half_t* q16     = (half_t*)take((size_t)B_ * N_ * C_ * 2);
  half_t* xk16    = (half_t*)take((size_t)B_ * M_ * C_ * 2);
  half_t* k16     = (half_t*)take((size_t)B_ * M_ * C_ * 2);
  half_t* vT16    = (half_t*)take((size_t)B_ * M_ * C_ * 2);
  half_t* outU16  = (half_t*)take((size_t)B_ * N_ * C_ * 2);
  float*  colsum  = (float*)take((size_t)B_ * NH_ * HD_ * 4);
  float*  s2part  = (float*)take((size_t)B_ * NH_ * 144 * 4);
  float*  rsArr   = (float*)take((size_t)B_ * NH_ * 4);
  float*  biascol = (float*)take((size_t)B_ * C_ * 4);

  k_prep_weights<<<dim3(1024), dim3(256), 0, stream>>>(w_q, w_kv, w_proj, wqT,
                                                       wkvT, wpT);
  k_transpose_x<<<dim3(72, 8, 8), dim3(32, 8), 0, stream>>>(x, xt16);
  k_qproj<<<dim3(1152, 4), dim3(128), 0, stream>>>(xt16, wqT, q16);
  k_sr_ln<<<dim3(B_ * M_), dim3(256), 0, stream>>>(x, w_sr, b_sr, g_ln, b_ln,
                                                   xk16);
  k_kvproj<<<dim3(288, 8), dim3(128), 0, stream>>>(xk16, wkvT, k16, vT16);
  k_vsum<<<dim3(B_), dim3(256), 0, stream>>>(vT16, colsum);
  k_attn<<<dim3(64, 36), dim3(128), 0, stream>>>(q16, k16, vT16, w_tc, b_tc,
                                                 outU16, s2part);
  k_norm<<<dim3(B_), dim3(256), 0, stream>>>(s2part, colsum, rsArr, biascol);
  k_proj<<<dim3(1152, 4), dim3(128), 0, stream>>>(outU16, wpT, rsArr, biascol,
                                                  b_proj, out);
}
